// SwinTransformerBlock_26190710571075
// MI455X (gfx1250) — matrix operand order repaired and checked by reading, not compiled, not run on hardware
//
#include <hip/hip_runtime.h>
#include <hip/hip_bf16.h>

typedef __attribute__((ext_vector_type(16))) _Float16 v16h;
typedef __attribute__((ext_vector_type(4)))  _Float16 v4h;
typedef __attribute__((ext_vector_type(8)))  float    v8f;
typedef __attribute__((ext_vector_type(4)))  int      v4i;

#define WMMA_F32_F16(a, b, c) \
  __builtin_amdgcn_wmma_f32_16x16x32_f16(false, (a), false, (b), (short)0, (c), false, false)

#define USE_ASYNC_LDS \
  (__has_builtin(__builtin_amdgcn_global_load_async_to_lds_b128) && \
   __has_builtin(__builtin_amdgcn_s_wait_asynccnt))

// ---------------- problem constants ----------------
// B=16, H=W=128, C=128, Wh=Ww=8, SH=4, NH=4, HD=32, N=64, HID=512
static constexpr float kEPS   = 1e-3f;
static constexpr float kSCALE = 0.17677669529663687f;  // 32^-0.5

// ---------------- WMMA fragment loaders (CDNA5 wave32 layouts) ----------------
// A: 16x32 f16, row-major source with leading dim `ld` (k = 16*(lane/16) + s: one contiguous 32B run/lane).
__device__ inline v16h load_A(const _Float16* p, int ld, int lane) {
  const int m  = lane & 15;
  const int kb = (lane < 16) ? 0 : 16;
  v16h a;
#pragma unroll
  for (int i = 0; i < 8; ++i) {
    const int k = kb + 2 * i;
    a[2 * i]     = p[m * ld + k];
    a[2 * i + 1] = p[m * ld + k + 1];
  }
  return a;
}

// B fragment from swizzled storage: one contiguous 32B load per lane.
// Swizzled tile (512 halfs): half j of lane l = B[kt*32 + (l<16?0:16) + j][nt*16 + (l&15)]
__device__ inline v16h load_B_sw(const _Float16* tile, int lane) {
  return *(const v16h*)(tile + lane * 16);
}

// ---------------- weight f32 -> f16 + B-fragment swizzle ----------------
// source: row-major K x N (K mult of 32, N mult of 16)
__global__ void swz_f16_kernel(const float* __restrict__ s, _Float16* __restrict__ d,
                               int K, int N) {
  const int idx = blockIdx.x * 256 + threadIdx.x;
  if (idx >= K * N) return;
  const int tile = idx >> 9;  // 512 halfs per tile
  const int rem  = idx & 511;
  const int l = rem >> 4, j = rem & 15;
  const int ntiles = N >> 4;
  const int kt = tile / ntiles, nt = tile % ntiles;
  const int k = kt * 32 + ((l < 16) ? 0 : 16) + j;
  const int n = nt * 16 + (l & 15);
  d[idx] = (_Float16)s[k * N + n];
}

// ---------------- combined rel-pos bias + window mask ----------------
// cbias[wi][h][m][n], wi<256, h<4, (m,n)<64x64  -> 4M floats (16MB, L2 resident)
__global__ void bias_combine_kernel(const float* __restrict__ bias_table,
                                    const int* __restrict__ rel_index,
                                    const float* __restrict__ attn_mask,
                                    float* __restrict__ cbias) {
  const int idx = blockIdx.x * 256 + threadIdx.x;  // 4,194,304 total
  const int mn = idx & 4095;
  const int h  = (idx >> 12) & 3;
  const int wi = idx >> 14;
  cbias[idx] = bias_table[rel_index[mn] * 4 + h] + attn_mask[wi * 4096 + mn];
}

// ---------------- fused windowed attention + proj + LN + residual ----------------
// grid: 4096 (= B * 256 windows), block: 256 threads (8 waves)
__global__ __launch_bounds__(256) void swin_attn_kernel(
    const float* __restrict__ x,
    const _Float16* __restrict__ w_qkvh,   // swizzled (128,384)
    const float* __restrict__ b_qkv,
    const _Float16* __restrict__ w_projh,  // swizzled (128,128)
    const float* __restrict__ b_proj,
    const float* __restrict__ g1, const float* __restrict__ be1,
    const float* __restrict__ cbias,       // (256,4,64,64)
    float* __restrict__ xout,
    _Float16* __restrict__ x1h)
{
  __shared__ float smemA[13312];                       // 53248 bytes
  _Float16* sXW = (_Float16*)smemA;                    // [0, 8192): 64x128 shifted window
  _Float16* sQ  = (_Float16*)smemA + 8192;             // [8192,10240): 64x32 q (row-major)
  _Float16* sKT = (_Float16*)smemA + 10240;            // [10240,12288): K^T B-frag swizzled
  _Float16* sVs = (_Float16*)smemA + 12288;            // [12288,14336): V B-frag swizzled
  _Float16* sP  = (_Float16*)smemA + 14336;            // [14336,18432): 64x64 probs
  _Float16* sAO = (_Float16*)smemA + 18432;            // [18432,26624): 64x128 attn output
  float*    sOutF = smemA;                             // 64x128 f32 proj out (reuse)

  const int lane = threadIdx.x & 31;
  const int wave = __builtin_amdgcn_readfirstlane((int)(threadIdx.x >> 5));  // scalar
  const int nlo  = lane & 15;
  const int ml   = (lane < 16) ? 0 : 8;

  const int b  = blockIdx.x >> 8;
  const int wi = blockIdx.x & 255;
  const int wy = wi >> 4, wx = wi & 15;

  // ---- phase 0: gather shifted window tokens -> LDS f16 ----
#pragma unroll
  for (int it = 0; it < 8; ++it) {
    const int idx4 = it * 256 + threadIdx.x;
    const int t  = idx4 >> 5;
    const int c4 = (idx4 & 31) << 2;
    const int r = t >> 3, c = t & 7;
    const int hy = (wy * 8 + r + 4) & 127;
    const int hx = (wx * 8 + c + 4) & 127;
    const float4 v = *(const float4*)(x + (((size_t)b * 16384 + hy * 128 + hx) << 7) + c4);
    const v4h hv = {(_Float16)v.x, (_Float16)v.y, (_Float16)v.z, (_Float16)v.w};
    *(v4h*)(sXW + t * 128 + c4) = hv;
  }
  __syncthreads();

  const v8f vzero = {0.f, 0.f, 0.f, 0.f, 0.f, 0.f, 0.f, 0.f};

  // ---- per-head attention ----
  for (int h = 0; h < 4; ++h) {
    // phase 1: qkv_h = XW @ w_qkv[:, head cols]; q -> row-major, k/v -> B-frag swizzled
#pragma unroll
    for (int q = 0; q < 3; ++q) {
      const int tt = wave + 8 * q;          // scalar
      const int rt = tt / 6, ct = tt % 6;   // scalar div
      const int part = ct >> 1;             // 0=q 1=k 2=v
      const int colbase = part * 128 + h * 32 + (ct & 1) * 16;
      v8f acc = vzero;
#pragma unroll
      for (int k4 = 0; k4 < 4; ++k4) {
        v16h a  = load_A(sXW + rt * 16 * 128 + 32 * k4, 128, lane);
        v16h bm = load_B_sw(w_qkvh + (size_t)(k4 * 24 + (colbase >> 4)) * 512, lane);
        acc = WMMA_F32_F16(a, bm, acc);
      }
      const float bb = b_qkv[colbase + nlo];
      if (part == 0) {
#pragma unroll
        for (int i = 0; i < 8; ++i)
          sQ[(rt * 16 + ml + i) * 32 + (ct & 1) * 16 + nlo] =
              (_Float16)((acc[i] + bb) * kSCALE);
      } else if (part == 1) {
        // element (token m, d): sKT tile rt, lane' = (m&15)+(d>=16?16:0), j = d&15
        const int loff = (ct & 1) ? 16 : 0;
#pragma unroll
        for (int i = 0; i < 8; ++i)
          sKT[rt * 512 + (ml + i + loff) * 16 + nlo] = (_Float16)(acc[i] + bb);
      } else {
        // element (token m, d): sVs tile (m>>5)*2 + (d>>4), lane' = (d&15)+((m>>4)&1)*16, j = m&15
        const int nt2 = ct & 1;
        const int kt2 = rt >> 1, khalf = rt & 1;
#pragma unroll
        for (int i = 0; i < 8; ++i)
          sVs[(kt2 * 2 + nt2) * 512 + (nlo + khalf * 16) * 16 + (ml + i)] =
              (_Float16)(acc[i] + bb);
      }
    }
    __syncthreads();

    // phase 2: scores = q @ k^T + cbias, softmax. waves 0-3 (scalar branch), one row-tile each.
    if (wave < 4) {
      const int rt = wave;
      const v16h aq = load_A(sQ + rt * 16 * 32, 32, lane);
      const float* cb = cbias + ((size_t)wi * 4 + h) * 4096;
      v8f sc[4];
#pragma unroll
      for (int nt = 0; nt < 4; ++nt) {
        v16h bk = load_B_sw(sKT + nt * 512, lane);
        sc[nt] = WMMA_F32_F16(aq, bk, vzero);
      }
#pragma unroll
      for (int nt = 0; nt < 4; ++nt)
#pragma unroll
        for (int i = 0; i < 8; ++i)
          sc[nt][i] += cb[(rt * 16 + ml + i) * 64 + nt * 16 + nlo];
      // row softmax: row = fixed VGPR i, spread over 4 frags x 16 lanes (half-wave)
#pragma unroll
      for (int i = 0; i < 8; ++i) {
        float mx = fmaxf(fmaxf(sc[0][i], sc[1][i]), fmaxf(sc[2][i], sc[3][i]));
        for (int d = 1; d < 16; d <<= 1) mx = fmaxf(mx, __shfl_xor(mx, d, 32));
        float e[4], s = 0.f;
#pragma unroll
        for (int nt = 0; nt < 4; ++nt) { e[nt] = __expf(sc[nt][i] - mx); s += e[nt]; }
        for (int d = 1; d < 16; d <<= 1) s += __shfl_xor(s, d, 32);
        const float inv = 1.0f / s;
        const int m = rt * 16 + ml + i;
#pragma unroll
        for (int nt = 0; nt < 4; ++nt)
          sP[m * 64 + nt * 16 + nlo] = (_Float16)(e[nt] * inv);
      }
    }
    __syncthreads();

    // phase 3: out_h = P @ V (64x32), one 16x16 tile per wave
    {
      const int rt = wave >> 1, ct = wave & 1;  // scalar
      v8f acc = vzero;
#pragma unroll
      for (int ks = 0; ks < 2; ++ks) {
        v16h ap = load_A(sP + rt * 16 * 64 + 32 * ks, 64, lane);
        v16h bv = load_B_sw(sVs + (ks * 2 + ct) * 512, lane);
        acc = WMMA_F32_F16(ap, bv, acc);
      }
#pragma unroll
      for (int i = 0; i < 8; ++i)
        sAO[(rt * 16 + ml + i) * 128 + h * 32 + ct * 16 + nlo] = (_Float16)acc[i];
    }
    __syncthreads();
  }

  // ---- phase 4: proj = AO @ w_proj + b_proj -> sOutF (f32) ----
  if (lane == 0) __builtin_prefetch(w_projh, 0, 1);
#pragma unroll
  for (int q = 0; q < 4; ++q) {
    const int tt = wave + 8 * q;              // scalar
    const int rt = tt >> 3, ct = tt & 7;
    v8f acc = vzero;
#pragma unroll
    for (int k4 = 0; k4 < 4; ++k4) {
      v16h a  = load_A(sAO + rt * 16 * 128 + 32 * k4, 128, lane);
      v16h bm = load_B_sw(w_projh + (size_t)(k4 * 8 + ct) * 512, lane);
      acc = WMMA_F32_F16(a, bm, acc);
    }
    const float bb = b_proj[ct * 16 + nlo];
#pragma unroll
    for (int i = 0; i < 8; ++i)
      sOutF[(rt * 16 + ml + i) * 128 + ct * 16 + nlo] = acc[i] + bb;
  }
  __syncthreads();

  // ---- phase 5: per-token LayerNorm + residual; scatter back to unshifted pixel ----
  {
    const int row = threadIdx.x >> 2;
    const int qd  = threadIdx.x & 3;
    const int cb0 = qd * 32;
    const float* orow = sOutF + row * 128;
    float s = 0.f, sq = 0.f;
#pragma unroll
    for (int j4 = 0; j4 < 8; ++j4) {
      const float4 v = *(const float4*)(orow + cb0 + 4 * j4);
      s  += v.x + v.y + v.z + v.w;
      sq += v.x * v.x + v.y * v.y + v.z * v.z + v.w * v.w;
    }
    s  += __shfl_xor(s, 1, 32);  s  += __shfl_xor(s, 2, 32);
    sq += __shfl_xor(sq, 1, 32); sq += __shfl_xor(sq, 2, 32);
    const float mean = s * (1.f / 128.f);
    const float var  = sq * (1.f / 128.f) - mean * mean;
    const float rstd = rsqrtf(var + kEPS);
    const int r = row >> 3, c = row & 7;
    const int hy = (wy * 8 + r + 4) & 127;
    const int hx = (wx * 8 + c + 4) & 127;
    const size_t base = ((size_t)b * 16384 + hy * 128 + hx) << 7;
#pragma unroll
    for (int j4 = 0; j4 < 8; ++j4) {
      const int ch = cb0 + 4 * j4;
      const float4 ov = *(const float4*)(orow + ch);
      const float4 xv = *(const float4*)(x + base + ch);
      const float4 gv = *(const float4*)(g1 + ch);
      const float4 bv = *(const float4*)(be1 + ch);
      float4 o;
      o.x = xv.x + (ov.x - mean) * rstd * gv.x + bv.x;
      o.y = xv.y + (ov.y - mean) * rstd * gv.y + bv.y;
      o.z = xv.z + (ov.z - mean) * rstd * gv.z + bv.z;
      o.w = xv.w + (ov.w - mean) * rstd * gv.w + bv.w;
      *(float4*)(xout + base + ch) = o;
      const v4h hv = {(_Float16)o.x, (_Float16)o.y, (_Float16)o.z, (_Float16)o.w};
      *(v4h*)(x1h + base + ch) = hv;
    }
  }
}

// ---------------- fused MLP (fc1 + gelu + fc2) + LN + residual ----------------
// grid: 4096 (64-token tiles), block: 256 threads (8 waves)
__global__ __launch_bounds__(256) void swin_mlp_kernel(
    const _Float16* __restrict__ x1h,
    const _Float16* __restrict__ w_fc1h,  // swizzled (128,512)
    const float* __restrict__ b_fc1,
    const _Float16* __restrict__ w_fc2h,  // swizzled (512,128)
    const float* __restrict__ b_fc2,
    const float* __restrict__ g2, const float* __restrict__ be2,
    float* __restrict__ xio)
{
  __shared__ float smemM[8192];                     // 32 KB
  _Float16* sX1 = (_Float16*)smemM;                 // 64x128 f16
  _Float16* sH  = (_Float16*)(smemM + 4096);        // 64x128 f16 (hidden chunk)
  float*    sO  = smemM;                            // 64x128 f32 (reuse at end)

  const int lane = threadIdx.x & 31;
  const int wave = __builtin_amdgcn_readfirstlane((int)(threadIdx.x >> 5));  // scalar
  const int nlo  = lane & 15;
  const int ml   = (lane < 16) ? 0 : 8;
  const size_t R0 = (size_t)blockIdx.x * 64;

  // load 64x128 x1 tile (f16) -> LDS
#if USE_ASYNC_LDS
  {
    const _Float16* src = x1h + R0 * 128;
#pragma unroll
    for (int q = 0; q < 4; ++q) {
      const int i = q * 256 + threadIdx.x;  // 1024 x 16B transfers
      __builtin_amdgcn_global_load_async_to_lds_b128(
          (__attribute__((address_space(1))) v4i*)(src + i * 8),
          (__attribute__((address_space(3))) v4i*)(sX1 + i * 8), 0, 0);
    }
    __builtin_amdgcn_s_wait_asynccnt(0);
  }
#else
  {
    const uint4* src = (const uint4*)(x1h + R0 * 128);
    uint4* dst = (uint4*)sX1;
    for (int i = threadIdx.x; i < 1024; i += 256) dst[i] = src[i];
  }
#endif
  __syncthreads();

  const v8f vzero = {0.f, 0.f, 0.f, 0.f, 0.f, 0.f, 0.f, 0.f};
  v8f acc2[4];
#pragma unroll
  for (int i = 0; i < 4; ++i) acc2[i] = vzero;

  for (int cc = 0; cc < 4; ++cc) {  // hidden dim in 4 chunks of 128
    if (cc < 3 && lane == 0)
      __builtin_prefetch(w_fc1h + (size_t)(cc + 1) * 8 * 512, 0, 1);

    // fc1 + exact gelu -> sH
#pragma unroll
    for (int q = 0; q < 4; ++q) {
      const int tt = wave + 8 * q;          // scalar
      const int rt = tt >> 3, ct = tt & 7;
      v8f acc = vzero;
#pragma unroll
      for (int k4 = 0; k4 < 4; ++k4) {
        v16h a  = load_A(sX1 + rt * 16 * 128 + 32 * k4, 128, lane);
        v16h bm = load_B_sw(w_fc1h + (size_t)(k4 * 32 + cc * 8 + ct) * 512, lane);
        acc = WMMA_F32_F16(a, bm, acc);
      }
      const float bb = b_fc1[cc * 128 + ct * 16 + nlo];
#pragma unroll
      for (int i = 0; i < 8; ++i) {
        const float v = acc[i] + bb;
        const float g = 0.5f * v * (1.0f + erff(v * 0.70710678118654752f));
        sH[(rt * 16 + ml + i) * 128 + ct * 16 + nlo] = (_Float16)g;
      }
    }
    __syncthreads();

    // fc2 partial accumulation (accumulators persist across chunks)
#pragma unroll
    for (int i = 0; i < 4; ++i) {
      const int tt = wave + 8 * i;          // scalar
      const int rt = tt >> 3, ct = tt & 7;
#pragma unroll
      for (int k4 = 0; k4 < 4; ++k4) {
        v16h a  = load_A(sH + rt * 16 * 128 + 32 * k4, 128, lane);
        v16h bm = load_B_sw(w_fc2h + (size_t)((cc * 4 + k4) * 8 + ct) * 512, lane);
        acc2[i] = WMMA_F32_F16(a, bm, acc2[i]);
      }
    }
    __syncthreads();
  }

  // write fc2 result (f32) -> sO
#pragma unroll
  for (int i = 0; i < 4; ++i) {
    const int tt = wave + 8 * i;
    const int rt = tt >> 3, ct = tt & 7;
    const float bb = b_fc2[ct * 16 + nlo];
#pragma unroll
    for (int j = 0; j < 8; ++j)
      sO[(rt * 16 + ml + j) * 128 + ct * 16 + nlo] = acc2[i][j] + bb;
  }
  __syncthreads();

  // LN + residual
  {
    const int row = threadIdx.x >> 2;
    const int qd  = threadIdx.x & 3;
    const int cb0 = qd * 32;
    const float* orow = sO + row * 128;
    float s = 0.f, sq = 0.f;
#pragma unroll
    for (int j4 = 0; j4 < 8; ++j4) {
      const float4 v = *(const float4*)(orow + cb0 + 4 * j4);
      s  += v.x + v.y + v.z + v.w;
      sq += v.x * v.x + v.y * v.y + v.z * v.z + v.w * v.w;
    }
    s  += __shfl_xor(s, 1, 32);  s  += __shfl_xor(s, 2, 32);
    sq += __shfl_xor(sq, 1, 32); sq += __shfl_xor(sq, 2, 32);
    const float mean = s * (1.f / 128.f);
    const float var  = sq * (1.f / 128.f) - mean * mean;
    const float rstd = rsqrtf(var + kEPS);
    const size_t base = (R0 + row) * 128;
#pragma unroll
    for (int j4 = 0; j4 < 8; ++j4) {
      const int ch = cb0 + 4 * j4;
      const float4 ov = *(const float4*)(orow + ch);
      const float4 xv = *(const float4*)(xio + base + ch);
      const float4 gv = *(const float4*)(g2 + ch);
      const float4 bv = *(const float4*)(be2 + ch);
      float4 o;
      o.x = xv.x + (ov.x - mean) * rstd * gv.x + bv.x;
      o.y = xv.y + (ov.y - mean) * rstd * gv.y + bv.y;
      o.z = xv.z + (ov.z - mean) * rstd * gv.z + bv.z;
      o.w = xv.w + (ov.w - mean) * rstd * gv.w + bv.w;
      *(float4*)(xio + base + ch) = o;
    }
  }
}

// ---------------- host launcher ----------------
extern "C" void kernel_launch(void* const* d_in, const int* in_sizes, int n_in,
                              void* d_out, int out_size, void* d_ws, size_t ws_size,
                              hipStream_t stream) {
  const float* x         = (const float*)d_in[0];
  const float* w_qkv     = (const float*)d_in[1];
  const float* b_qkv     = (const float*)d_in[2];
  const float* bias_tab  = (const float*)d_in[3];
  const float* w_proj    = (const float*)d_in[4];
  const float* b_proj    = (const float*)d_in[5];
  const float* g1        = (const float*)d_in[6];
  const float* be1       = (const float*)d_in[7];
  const float* w_fc1     = (const float*)d_in[8];
  const float* b_fc1     = (const float*)d_in[9];
  const float* w_fc2     = (const float*)d_in[10];
  const float* b_fc2     = (const float*)d_in[11];
  const float* g2        = (const float*)d_in[12];
  const float* be2       = (const float*)d_in[13];
  const int*   rel_index = (const int*)d_in[14];
  const float* attn_mask = (const float*)d_in[15];
  float* xout = (float*)d_out;

  // workspace layout (halfs)
  _Float16* wsh     = (_Float16*)d_ws;
  _Float16* w_qkvh  = wsh;               // 49152 (swizzled)
  _Float16* w_projh = wsh + 49152;       // 16384 (swizzled)
  _Float16* w_fc1h  = wsh + 65536;       // 65536 (swizzled)
  _Float16* w_fc2h  = wsh + 131072;      // 65536 (swizzled)
  _Float16* x1h     = wsh + 196608;      // 33554432
  float*    cbias   = (float*)(wsh + 33751040);  // 4,194,304 floats (16MB)

  swz_f16_kernel<<<(49152 + 255) / 256, 256, 0, stream>>>(w_qkv, w_qkvh, 128, 384);
  swz_f16_kernel<<<(16384 + 255) / 256, 256, 0, stream>>>(w_proj, w_projh, 128, 128);
  swz_f16_kernel<<<(65536 + 255) / 256, 256, 0, stream>>>(w_fc1, w_fc1h, 128, 512);
  swz_f16_kernel<<<(65536 + 255) / 256, 256, 0, stream>>>(w_fc2, w_fc2h, 512, 128);
  bias_combine_kernel<<<16384, 256, 0, stream>>>(bias_tab, rel_index, attn_mask, cbias);

  swin_attn_kernel<<<4096, 256, 0, stream>>>(x, w_qkvh, b_qkv, w_projh, b_proj,
                                             g1, be1, cbias, xout, x1h);
  swin_mlp_kernel<<<4096, 256, 0, stream>>>(x1h, w_fc1h, b_fc1, w_fc2h, b_fc2, g2, be2, xout);
}
